// AEN_7189775253647
// MI455X (gfx1250) — matrix operand order repaired and checked by reading, not compiled, not run on hardware
//
#include <hip/hip_runtime.h>
#include <math.h>

#define DMODEL 1024
#define ODIM   512
#define NROWS  8192

typedef __attribute__((ext_vector_type(16))) __bf16 v16bf;
typedef __attribute__((ext_vector_type(8)))  __bf16 v8bf;
typedef __attribute__((ext_vector_type(8)))  float  v8f;
typedef __attribute__((ext_vector_type(4)))  float  v4f;

// ---------------------------------------------------------------------------
// Kernel 1: W [O,D] f32  ->  WT [D,O] bf16  (so WMMA B-operand loads are
// coalesced: 16 lanes read 32B per k row)
// ---------------------------------------------------------------------------
__global__ __launch_bounds__(256)
void wt_kernel(const float* __restrict__ W, __bf16* __restrict__ WT) {
    int idx = blockIdx.x * 256 + threadIdx.x;
    if (idx >= ODIM * DMODEL) return;
    int o = idx / DMODEL;
    int d = idx - o * DMODEL;
    WT[(size_t)d * ODIM + o] = (__bf16)W[idx];
}

// ---------------------------------------------------------------------------
// Kernel 2: projection  P = X @ WT + b  (optional LayerNorm over O=512)
//   MODE 0: f32 row-major out        (queries V-proj -> q_prototype)
//   MODE 1: bf16 row-major out       (support V-proj -> Vs)
//   MODE 2: bf16 row-major out, LN   (queries Q-proj -> Q_ln)
//   MODE 3: bf16 TRANSPOSED out, LN  (support K-proj -> K_lnT [512][8192])
// Block = 16 rows x 512 cols; 8 waves, each wave owns 4 N-tiles of 16.
// ---------------------------------------------------------------------------
template<int MODE>
__global__ __launch_bounds__(256)
void proj_kernel(const float* __restrict__ X, const __bf16* __restrict__ WT,
                 const float* __restrict__ bias,
                 const float* __restrict__ gamma, const float* __restrict__ beta,
                 float* __restrict__ outf, __bf16* __restrict__ outb)
{
    const int wave = threadIdx.x >> 5;
    const int lane = threadIdx.x & 31;
    const int h    = lane >> 4;      // K-half selector for A operand
    const int r    = lane & 15;      // row (A) / column (C)
    const int m0   = blockIdx.x * 16;
    const int n0   = wave * 64;

    __shared__ float s_sum[16], s_sq[16];

    v8f acc[4];
#pragma unroll
    for (int t = 0; t < 4; ++t)
#pragma unroll
        for (int v = 0; v < 8; ++v) acc[t][v] = 0.0f;

    const float* xrow = X + (size_t)(m0 + r) * DMODEL;

    for (int k = 0; k < DMODEL; k += 32) {
        // A tile 16x32 bf16: lane holds row m0+r, K chunks [k+8h,+8) and [k+16+8h,+8)
        v16bf a;
        {
            const float* p0 = xrow + k + 8 * h;
            const float* p1 = xrow + k + 16 + 8 * h;
            v4f f0 = *(const v4f*)(p0);
            v4f f1 = *(const v4f*)(p0 + 4);
            v4f f2 = *(const v4f*)(p1);
            v4f f3 = *(const v4f*)(p1 + 4);
#pragma unroll
            for (int i = 0; i < 4; ++i) {
                a[i]      = (__bf16)f0[i];
                a[4 + i]  = (__bf16)f1[i];
                a[8 + i]  = (__bf16)f2[i];
                a[12 + i] = (__bf16)f3[i];
            }
        }
        // B tiles 32x16 bf16: lane holds column N=n0+t*16+r, K chunks [k+8h,+8) and [k+16+8h,+8)
        const __bf16* bp = WT + (size_t)k * ODIM + n0 + r;
#pragma unroll
        for (int t = 0; t < 4; ++t) {
            v16bf b;
#pragma unroll
            for (int i = 0; i < 8; ++i) {
                b[i]     = bp[(size_t)(8 * h + i) * ODIM + t * 16];
                b[8 + i] = bp[(size_t)(16 + 8 * h + i) * ODIM + t * 16];
            }
            acc[t] = __builtin_amdgcn_wmma_f32_16x16x32_bf16(
                false, a, false, b, (short)0, acc[t], false, false);
        }
    }

    // fused bias (column-indexed; same for all 8 row-VGPRs)
#pragma unroll
    for (int t = 0; t < 4; ++t) {
        float bv = bias[n0 + t * 16 + r];
#pragma unroll
        for (int v = 0; v < 8; ++v) acc[t][v] += bv;
    }

    if (MODE >= 2) {
        if (threadIdx.x < 16) { s_sum[threadIdx.x] = 0.0f; s_sq[threadIdx.x] = 0.0f; }
        __syncthreads();
        float ps[8], pq[8];
#pragma unroll
        for (int v = 0; v < 8; ++v) {
            float s = 0.0f, q = 0.0f;
#pragma unroll
            for (int t = 0; t < 4; ++t) { float x = acc[t][v]; s += x; q += x * x; }
            ps[v] = s; pq[v] = q;
        }
        // reduce over the 16 lanes sharing the same rows
#pragma unroll
        for (int off = 1; off < 16; off <<= 1)
#pragma unroll
            for (int v = 0; v < 8; ++v) {
                ps[v] += __shfl_xor(ps[v], off, 32);
                pq[v] += __shfl_xor(pq[v], off, 32);
            }
        if (r == 0) {
#pragma unroll
            for (int v = 0; v < 8; ++v) {
                atomicAdd(&s_sum[8 * h + v], ps[v]);
                atomicAdd(&s_sq[8 * h + v], pq[v]);
            }
        }
        __syncthreads();
#pragma unroll
        for (int v = 0; v < 8; ++v) {
            int row = 8 * h + v;
            float mu  = s_sum[row] * (1.0f / ODIM);
            float var = s_sq[row] * (1.0f / ODIM) - mu * mu;
            float inv = rsqrtf(var + 1e-5f);
#pragma unroll
            for (int t = 0; t < 4; ++t) {
                int n = n0 + t * 16 + r;
                acc[t][v] = (acc[t][v] - mu) * inv * gamma[n] + beta[n];
            }
        }
    }

    // store
#pragma unroll
    for (int t = 0; t < 4; ++t) {
#pragma unroll
        for (int v = 0; v < 8; ++v) {
            int row = m0 + 8 * h + v;
            int n   = n0 + t * 16 + r;
            if (MODE == 0) {
                outf[(size_t)row * ODIM + n] = acc[t][v];
            } else if (MODE == 3) {
                outb[(size_t)n * NROWS + row] = (__bf16)acc[t][v];
            } else {
                outb[(size_t)row * ODIM + n] = (__bf16)acc[t][v];
            }
        }
    }
}

// ---------------------------------------------------------------------------
// Kernel 3: fused flash attention.
//   out[q,:] = softmax_s(Q_ln[q]·K_ln[s] / sqrt(512)) @ Vs
// Block = 16 queries; 8 waves. Per 128-support chunk: each wave computes one
// 16x16 score tile (16 WMMAs over K=512), online softmax via LDS stats,
// probabilities staged in LDS as bf16, then 4x4 = 16 PV WMMAs per wave.
// ---------------------------------------------------------------------------
__global__ __launch_bounds__(256)
void attn_kernel(const __bf16* __restrict__ Q, const __bf16* __restrict__ KT,
                 const __bf16* __restrict__ V, float* __restrict__ out)
{
    constexpr int PPAD = 136;                  // bf16 elements per prob row (+pad)
    __shared__ __bf16 s_p[16 * PPAD];
    __shared__ float s_wmax[8][16], s_wsum[8][16];
    __shared__ float s_m[16], s_l[16], s_f[16];

    const int wave = threadIdx.x >> 5;
    const int lane = threadIdx.x & 31;
    const int h    = lane >> 4;
    const int r    = lane & 15;
    const int q0   = blockIdx.x * 16;

    // preload all Q A-operands for this block's 16 rows (K = 0..511)
    v16bf aq[16];
    {
        const __bf16* qrow = Q + (size_t)(q0 + r) * ODIM;
#pragma unroll
        for (int kk = 0; kk < 16; ++kk) {
            v8bf lo = *(const v8bf*)(qrow + kk * 32 + 8 * h);
            v8bf hi = *(const v8bf*)(qrow + kk * 32 + 16 + 8 * h);
#pragma unroll
            for (int i = 0; i < 8; ++i) { aq[kk][i] = lo[i]; aq[kk][8 + i] = hi[i]; }
        }
    }

    v8f acc[4];
#pragma unroll
    for (int t = 0; t < 4; ++t)
#pragma unroll
        for (int v = 0; v < 8; ++v) acc[t][v] = 0.0f;

    if (threadIdx.x < 16) { s_m[threadIdx.x] = -INFINITY; s_l[threadIdx.x] = 0.0f; }
    __syncthreads();

    const float sscale = 0.044194173824159216f;   // 1/sqrt(512)

    for (int s0 = 0; s0 < NROWS; s0 += 128) {
        // ---- scores: this wave's 16x16 tile at support cols s0 + wave*16 ----
        v8f s;
#pragma unroll
        for (int v = 0; v < 8; ++v) s[v] = 0.0f;
        {
            const __bf16* kp = KT + (size_t)(s0 + wave * 16 + r);
#pragma unroll
            for (int kk = 0; kk < 16; ++kk) {
                v16bf b;
#pragma unroll
                for (int i = 0; i < 8; ++i) {
                    b[i]     = kp[(size_t)(kk * 32 + 8 * h + i) * NROWS];
                    b[8 + i] = kp[(size_t)(kk * 32 + 16 + 8 * h + i) * NROWS];
                }
                s = __builtin_amdgcn_wmma_f32_16x16x32_bf16(
                    false, aq[kk], false, b, (short)0, s, false, false);
            }
        }
#pragma unroll
        for (int v = 0; v < 8; ++v) s[v] *= sscale;

        // ---- wave-local row maxima ----
        float pm[8];
#pragma unroll
        for (int v = 0; v < 8; ++v) pm[v] = s[v];
#pragma unroll
        for (int off = 1; off < 16; off <<= 1)
#pragma unroll
            for (int v = 0; v < 8; ++v) pm[v] = fmaxf(pm[v], __shfl_xor(pm[v], off, 32));
        if (r == 0)
#pragma unroll
            for (int v = 0; v < 8; ++v) s_wmax[wave][8 * h + v] = pm[v];
        __syncthreads();

        // ---- combine maxima, update running stats ----
        if (threadIdx.x < 16) {
            int row = threadIdx.x;
            float cm = s_wmax[0][row];
#pragma unroll
            for (int w = 1; w < 8; ++w) cm = fmaxf(cm, s_wmax[w][row]);
            float mo = s_m[row];
            float mn = fmaxf(mo, cm);
            float fac = __expf(mo - mn);      // 0 when mo == -inf
            s_f[row] = fac;
            s_m[row] = mn;
            s_l[row] *= fac;
        }
        __syncthreads();

        // ---- probabilities (from registers) + wave-local row sums ----
        float ps[8];
#pragma unroll
        for (int v = 0; v < 8; ++v) {
            int row = 8 * h + v;
            float p = __expf(s[v] - s_m[row]);
            s_p[row * PPAD + wave * 16 + r] = (__bf16)p;
            ps[v] = p;
        }
#pragma unroll
        for (int off = 1; off < 16; off <<= 1)
#pragma unroll
            for (int v = 0; v < 8; ++v) ps[v] += __shfl_xor(ps[v], off, 32);
        if (r == 0)
#pragma unroll
            for (int v = 0; v < 8; ++v) s_wsum[wave][8 * h + v] = ps[v];

        // ---- rescale output accumulator by exp(m_old - m_new) ----
#pragma unroll
        for (int v = 0; v < 8; ++v) {
            float fac = s_f[8 * h + v];
#pragma unroll
            for (int t = 0; t < 4; ++t) acc[t][v] *= fac;
        }
        __syncthreads();

        if (threadIdx.x < 16) {
            float add = 0.0f;
#pragma unroll
            for (int w = 0; w < 8; ++w) add += s_wsum[w][threadIdx.x];
            s_l[threadIdx.x] += add;
        }

        // ---- PV accumulation: P(16x128) @ V(128x512) ----
#pragma unroll
        for (int kc = 0; kc < 4; ++kc) {
            v16bf a;
            const __bf16* pp = &s_p[r * PPAD + kc * 32];
            v8bf lo = *(const v8bf*)(pp + 8 * h);
            v8bf hi = *(const v8bf*)(pp + 16 + 8 * h);
#pragma unroll
            for (int i = 0; i < 8; ++i) { a[i] = lo[i]; a[8 + i] = hi[i]; }
            const __bf16* vp = V + (size_t)(s0 + kc * 32) * ODIM + wave * 64 + r;
#pragma unroll
            for (int t = 0; t < 4; ++t) {
                v16bf b;
#pragma unroll
                for (int i = 0; i < 8; ++i) {
                    b[i]     = vp[(size_t)(8 * h + i) * ODIM + t * 16];
                    b[8 + i] = vp[(size_t)(16 + 8 * h + i) * ODIM + t * 16];
                }
                acc[t] = __builtin_amdgcn_wmma_f32_16x16x32_bf16(
                    false, a, false, b, (short)0, acc[t], false, false);
            }
        }
        __syncthreads();
    }

    // ---- normalize by row sum and store ----
#pragma unroll
    for (int v = 0; v < 8; ++v) {
        int row = 8 * h + v;
        float invl = 1.0f / s_l[row];
#pragma unroll
        for (int t = 0; t < 4; ++t) {
            out[(size_t)(q0 + row) * ODIM + wave * 64 + t * 16 + r] = acc[t][v] * invl;
        }
    }
}

// ---------------------------------------------------------------------------
extern "C" void kernel_launch(void* const* d_in, const int* in_sizes, int n_in,
                              void* d_out, int out_size, void* d_ws, size_t ws_size,
                              hipStream_t stream)
{
    const float* support = (const float*)d_in[0];
    const float* queries = (const float*)d_in[1];
    const float* Wq = (const float*)d_in[2];
    const float* bq = (const float*)d_in[3];
    const float* Wk = (const float*)d_in[4];
    const float* bk = (const float*)d_in[5];
    const float* Wv = (const float*)d_in[6];
    const float* bv = (const float*)d_in[7];
    const float* ln_g = (const float*)d_in[8];
    const float* ln_b = (const float*)d_in[9];
    (void)in_sizes; (void)n_in; (void)out_size; (void)ws_size;

    float* out_q = (float*)d_out;                   // q_prototype  [8192,512]
    float* out_c = out_q + (size_t)NROWS * ODIM;    // class_proto  [8192,512]

    char* ws = (char*)d_ws;
    size_t off = 0;
    __bf16* WTq = (__bf16*)(ws + off); off += (size_t)DMODEL * ODIM * 2;
    __bf16* WTk = (__bf16*)(ws + off); off += (size_t)DMODEL * ODIM * 2;
    __bf16* WTv = (__bf16*)(ws + off); off += (size_t)DMODEL * ODIM * 2;
    __bf16* Qs  = (__bf16*)(ws + off); off += (size_t)NROWS * ODIM * 2;   // Q_ln  [8192,512]
    __bf16* KsT = (__bf16*)(ws + off); off += (size_t)NROWS * ODIM * 2;   // K_lnT [512,8192]
    __bf16* Vs  = (__bf16*)(ws + off); off += (size_t)NROWS * ODIM * 2;   // Vs    [8192,512]

    const int wt_blocks = (ODIM * DMODEL + 255) / 256;
    wt_kernel<<<wt_blocks, 256, 0, stream>>>(Wq, WTq);
    wt_kernel<<<wt_blocks, 256, 0, stream>>>(Wk, WTk);
    wt_kernel<<<wt_blocks, 256, 0, stream>>>(Wv, WTv);

    const int pj_blocks = NROWS / 16;
    // q_prototype = queries @ Wv.T + bv (f32 direct to output)
    proj_kernel<0><<<pj_blocks, 256, 0, stream>>>(queries, WTv, bv, nullptr, nullptr, out_q, nullptr);
    // support_vs (bf16 row-major)
    proj_kernel<1><<<pj_blocks, 256, 0, stream>>>(support, WTv, bv, nullptr, nullptr, nullptr, Vs);
    // queries_qs = LN(queries @ Wq.T + bq) (bf16 row-major)
    proj_kernel<2><<<pj_blocks, 256, 0, stream>>>(queries, WTq, bq, ln_g, ln_b, nullptr, Qs);
    // support_ks = LN(support @ Wk.T + bk) (bf16 TRANSPOSED [512,8192])
    proj_kernel<3><<<pj_blocks, 256, 0, stream>>>(support, WTk, bk, ln_g, ln_b, nullptr, KsT);

    // fused scores + softmax + PV
    attn_kernel<<<NROWS / 16, 256, 0, stream>>>(Qs, KsT, Vs, out_c);
}
